// SFE_13469017440360
// MI455X (gfx1250) — matrix operand order repaired and checked by reading, not compiled, not run on hardware
//
#include <hip/hip_runtime.h>
#include <math.h>

typedef __bf16 bf16;
typedef __attribute__((ext_vector_type(16))) __bf16 v16bf;
typedef __attribute__((ext_vector_type(8)))  __bf16 v8bf;
typedef __attribute__((ext_vector_type(8)))  float  v8f;

#define B_    4
#define C_    256
#define CC_   64
#define H_    64
#define W_    64
#define HW_   4096
#define NPIX_ 16384  // B*H*W

__device__ __forceinline__ float sigm(float v) { return 1.f / (1.f + expf(-v)); }

// A-tile K-permutation: aligned 8-groups {0,8,16,24} -> {0,16,8,24} so each lane
// half reads 16 contiguous halves in the ISA A-fragment order.
__device__ __forceinline__ int apos(int kl) {
  return (kl >= 8 && kl < 24) ? (kl ^ 24) : kl;
}

// ---------------- base = x + y, per-(b,c) mean (float4) ----------------
__global__ void k_base_mean(const float4* __restrict__ x, const float4* __restrict__ y,
                            float4* __restrict__ base, float* __restrict__ smean) {
  int bc = blockIdx.x;
  const float4* xp = x + (size_t)bc * (HW_ / 4);
  const float4* yp = y + (size_t)bc * (HW_ / 4);
  float4* bp = base + (size_t)bc * (HW_ / 4);
  float s = 0.f;
  for (int i = threadIdx.x; i < HW_ / 4; i += 256) {
    float4 a = xp[i], b = yp[i];
    float4 v = make_float4(a.x + b.x, a.y + b.y, a.z + b.z, a.w + b.w);
    bp[i] = v;
    s += v.x + v.y + v.z + v.w;
  }
  __shared__ float red[256];
  red[threadIdx.x] = s;
  __syncthreads();
  for (int off = 128; off > 0; off >>= 1) {
    if (threadIdx.x < off) red[threadIdx.x] += red[threadIdx.x + off];
    __syncthreads();
  }
  if (threadIdx.x == 0) smean[bc] = red[0] * (1.f / HW_);
}

// ---------------- SE MLP: onep[b,c] = 1 + sigmoid(relu(s@w1^T)@w2^T) ----------------
__global__ void k_se(const float* __restrict__ smean, const float* __restrict__ w1,
                     const float* __restrict__ w2, float* __restrict__ onep) {
  __shared__ float hid[B_][16];
  int t = threadIdx.x;
  if (t < 64) {
    int b = t >> 4, r = t & 15;
    float s = 0.f;
    for (int c = 0; c < C_; ++c) s += smean[b * C_ + c] * w1[r * C_ + c];
    hid[b][r] = fmaxf(s, 0.f);
  }
  __syncthreads();
  int c = t;
  for (int b = 0; b < B_; ++b) {
    float s = 0.f;
    #pragma unroll
    for (int r = 0; r < 16; ++r) s += hid[b][r] * w2[c * 16 + r];
    onep[b * C_ + c] = 1.f + sigm(s);
  }
}

// ---------------- base *= (1 + se_scale) (float4) ----------------
__global__ void k_scale(float4* __restrict__ base, const float* __restrict__ onep) {
  int i = blockIdx.x * 256 + threadIdx.x;
  float g = onep[i >> 10];
  float4 v = base[i];
  base[i] = make_float4(v.x * g, v.y * g, v.z * g, v.w * g);
}

// ---- weight relayout: src f32 [N][Cin*9] (k=c*9+r) -> dst bf16 [9][Npad][Cin] ----
__global__ void k_wprep(const float* __restrict__ src, bf16* __restrict__ dst,
                        int N, int Cin, int Npad) {
  int i = blockIdx.x * 256 + threadIdx.x;
  int total = 9 * Npad * Cin;
  if (i >= total) return;
  int r = i / (Npad * Cin);
  int rem = i - r * (Npad * Cin);
  int n = rem / Cin, c = rem - n * Cin;
  float v = (n < N) ? src[(size_t)n * Cin * 9 + c * 9 + r] : 0.f;
  dst[i] = (bf16)v;
}

// ---------------- offset conv 256->27 (3x3) as 9 shifted GEMMs, bf16 WMMA ----------------
__global__ void __launch_bounds__(128) k_conv_off(const float* __restrict__ base,
    const bf16* __restrict__ wR,      // [9][32][C_]
    const float* __restrict__ off_b, float* __restrict__ off) {
  __shared__ bf16 ldsA[32][32];
  __shared__ bf16 ldsBt[32][32];
  const int m0 = blockIdx.x * 32;
  const int tid = threadIdx.x;
  const int lane = tid & 31, wave = tid >> 5;
  const int wm = wave >> 1, wn = wave & 1;
  const int laneLo = lane & 15;
  const bool laneHi = lane >= 16;
  // A staging: row ml = tid>>2, 8 consecutive channels from acl0
  const int aml = tid >> 2, acl0 = (tid & 3) * 8, apr = acl0;
  const int am = m0 + aml, ab = am >> 12, ah = (am >> 6) & 63, aw = am & 63;
  // B staging: thread -> (n, 8 consecutive c)
  const int bnn = tid & 31, bj = (tid >> 5) * 8;

  v8f acc = {};
  for (int r = 0; r < 9; ++r) {
    const int yy = ah + r / 3 - 1, xx = aw + (r % 3) - 1;
    const bool valid = ((unsigned)yy < H_) && ((unsigned)xx < W_);
    const float* aptr = base + ((size_t)ab * C_ + acl0) * HW_ + yy * W_ + xx;
    const bf16* brow = wR + ((size_t)r * 32 + bnn) * C_ + bj;
    for (int c0 = 0; c0 < C_; c0 += 32) {
      v8bf av;
      if (valid) {
        const float* p = aptr + (size_t)c0 * HW_;
        #pragma unroll
        for (int j = 0; j < 8; ++j) { av[j] = (bf16)(*p); p += HW_; }
      } else {
        #pragma unroll
        for (int j = 0; j < 8; ++j) av[j] = (bf16)0.f;
      }
      *(v8bf*)&ldsA[aml][apr] = av;
      *(v8bf*)&ldsBt[bnn][bj] = *(const v8bf*)&brow[c0];
      __syncthreads();
      v16bf af = *(const v16bf*)&ldsA[wm * 16 + laneLo][laneHi ? 16 : 0];
      v16bf bf = *(const v16bf*)&ldsBt[wn * 16 + laneLo][laneHi ? 16 : 0];
      acc = __builtin_amdgcn_wmma_f32_16x16x32_bf16(false, af, false, bf, (short)0, acc, false, false);
      __syncthreads();
    }
  }
  int n = wn * 16 + laneLo;
  if (n < 27) {
    float bias = off_b[n];
    #pragma unroll
    for (int r = 0; r < 8; ++r) {
      int m = m0 + wm * 16 + r + (laneHi ? 8 : 0);
      off[(size_t)((m >> 12) * 27 + n) * HW_ + (m & 4095)] = acc[r] + bias;
    }
  }
}

// ---------------- DCNv2 sample + einsum -> enc (relu, bf16), 9 tap-GEMMs ----------------
__global__ void __launch_bounds__(128) k_dcn(const float* __restrict__ base,
    const float* __restrict__ off, const bf16* __restrict__ wR,   // [9][CC_][C_]
    const float* __restrict__ enc_b, bf16* __restrict__ enc) {
  __shared__ bf16 ldsA[32][32];
  __shared__ bf16 ldsBt[32][32];
  __shared__ int   tIdx[288][4];
  __shared__ float tW[288][4];
  const int m0 = blockIdx.x * 32;
  const int n0 = blockIdx.y * 32;
  const int tid = threadIdx.x;
  const int lane = tid & 31, wave = tid >> 5;
  const int wm = wave >> 1, wn = wave & 1;
  const int laneLo = lane & 15;
  const bool laneHi = lane >= 16;
  const int aml = tid >> 2, acl0 = (tid & 3) * 8, apr = acl0;
  const int am = m0 + aml, ab = am >> 12;
  const int bnn = tid & 31, bj = (tid >> 5) * 8;

  // Precompute bilinear taps (idx + weight*mask) per (pixel, tap); reuse over 256 ch.
  for (int t = tid; t < 288; t += 128) {
    int ml = t / 9, r = t - ml * 9;
    int m = m0 + ml, b = m >> 12, h = (m >> 6) & 63, w = m & 63;
    size_t ob = (size_t)(b * 27) * HW_ + h * W_ + w;
    float dy = off[ob + (size_t)(2 * r) * HW_];
    float dx = off[ob + (size_t)(2 * r + 1) * HW_];
    float mk = sigm(off[ob + (size_t)(18 + r) * HW_]);
    float py = dy + (float)h + (float)(r / 3 - 1);
    float px = dx + (float)w + (float)(r % 3 - 1);
    float y0 = floorf(py), x0 = floorf(px);
    float fy = py - y0, fx = px - x0;
    #pragma unroll
    for (int cnr = 0; cnr < 4; ++cnr) {
      float yc = y0 + (float)(cnr >> 1), xc = x0 + (float)(cnr & 1);
      float wy = (cnr >> 1) ? fy : 1.f - fy;
      float wx = (cnr & 1) ? fx : 1.f - fx;
      bool valid = (yc >= 0.f) && (yc < (float)H_) && (xc >= 0.f) && (xc < (float)W_);
      int iy = min(max((int)yc, 0), H_ - 1);
      int ix = min(max((int)xc, 0), W_ - 1);
      tIdx[t][cnr] = iy * W_ + ix;
      tW[t][cnr] = valid ? wy * wx * mk : 0.f;
    }
  }
  __syncthreads();

  const float* abase = base + ((size_t)ab * C_ + acl0) * HW_;
  v8f acc = {};
  for (int r = 0; r < 9; ++r) {
    const int te = aml * 9 + r;
    const int i0 = tIdx[te][0], i1 = tIdx[te][1], i2 = tIdx[te][2], i3 = tIdx[te][3];
    const float w0 = tW[te][0], w1 = tW[te][1], w2 = tW[te][2], w3 = tW[te][3];
    const bf16* brow = wR + ((size_t)r * CC_ + n0 + bnn) * C_ + bj;
    for (int c0 = 0; c0 < C_; c0 += 32) {
      v8bf av;
      const float* p = abase + (size_t)c0 * HW_;
      #pragma unroll
      for (int j = 0; j < 8; ++j) {
        av[j] = (bf16)(w0 * p[i0] + w1 * p[i1] + w2 * p[i2] + w3 * p[i3]);
        p += HW_;
      }
      *(v8bf*)&ldsA[aml][apr] = av;
      *(v8bf*)&ldsBt[bnn][bj] = *(const v8bf*)&brow[c0];
      __syncthreads();
      v16bf af = *(const v16bf*)&ldsA[wm * 16 + laneLo][laneHi ? 16 : 0];
      v16bf bf = *(const v16bf*)&ldsBt[wn * 16 + laneLo][laneHi ? 16 : 0];
      acc = __builtin_amdgcn_wmma_f32_16x16x32_bf16(false, af, false, bf, (short)0, acc, false, false);
      __syncthreads();
    }
  }
  int n = n0 + wn * 16 + laneLo;
  float bias = enc_b[n];
  #pragma unroll
  for (int r = 0; r < 8; ++r) {
    int m = m0 + wm * 16 + r + (laneHi ? 8 : 0);
    float v = fmaxf(acc[r] + bias, 0.f);
    enc[(size_t)((m >> 12) * CC_ + n) * HW_ + (m & 4095)] = (bf16)v;
  }
}

// ---------------- fused we/wf convs + BN + sigmoid + final gate ----------------
__global__ void __launch_bounds__(128) k_conv2(const bf16* __restrict__ enc,
    const bf16* __restrict__ weR, const bf16* __restrict__ wfR,   // [9][C_][CC_]
    const float* __restrict__ we_b, const float* __restrict__ we_g,
    const float* __restrict__ we_beta, const float* __restrict__ we_m,
    const float* __restrict__ we_v,
    const float* __restrict__ wf_b, const float* __restrict__ wf_g,
    const float* __restrict__ wf_beta, const float* __restrict__ wf_m,
    const float* __restrict__ wf_v,
    const float* __restrict__ x, const float* __restrict__ y,
    float* __restrict__ out) {
  __shared__ bf16 ldsA[32][32];
  __shared__ bf16 ldsBtE[32][32];
  __shared__ bf16 ldsBtF[32][32];
  const int m0 = blockIdx.x * 32;
  const int n0 = blockIdx.y * 32;
  const int tid = threadIdx.x;
  const int lane = tid & 31, wave = tid >> 5;
  const int wm = wave >> 1, wn = wave & 1;
  const int laneLo = lane & 15;
  const bool laneHi = lane >= 16;
  const int aml = tid >> 2, acl0 = (tid & 3) * 8, apr = acl0;
  const int am = m0 + aml, ab = am >> 12, ah = (am >> 6) & 63, aw = am & 63;
  const int bnn = tid & 31, bj = (tid >> 5) * 8;

  v8f accE = {}, accF = {};
  for (int r = 0; r < 9; ++r) {
    const int yy = ah + r / 3 - 1, xx = aw + (r % 3) - 1;
    const bool valid = ((unsigned)yy < H_) && ((unsigned)xx < W_);
    const bf16* aptr = enc + ((size_t)ab * CC_ + acl0) * HW_ + yy * W_ + xx;
    const bf16* browE = weR + ((size_t)r * C_ + n0 + bnn) * CC_ + bj;
    const bf16* browF = wfR + ((size_t)r * C_ + n0 + bnn) * CC_ + bj;
    for (int c0 = 0; c0 < CC_; c0 += 32) {
      v8bf av;
      if (valid) {
        const bf16* p = aptr + (size_t)c0 * HW_;
        #pragma unroll
        for (int j = 0; j < 8; ++j) { av[j] = *p; p += HW_; }
      } else {
        #pragma unroll
        for (int j = 0; j < 8; ++j) av[j] = (bf16)0.f;
      }
      *(v8bf*)&ldsA[aml][apr] = av;
      *(v8bf*)&ldsBtE[bnn][bj] = *(const v8bf*)&browE[c0];
      *(v8bf*)&ldsBtF[bnn][bj] = *(const v8bf*)&browF[c0];
      __syncthreads();
      v16bf af = *(const v16bf*)&ldsA[wm * 16 + laneLo][laneHi ? 16 : 0];
      v16bf be = *(const v16bf*)&ldsBtE[wn * 16 + laneLo][laneHi ? 16 : 0];
      v16bf bff = *(const v16bf*)&ldsBtF[wn * 16 + laneLo][laneHi ? 16 : 0];
      accE = __builtin_amdgcn_wmma_f32_16x16x32_bf16(false, af, false, be,  (short)0, accE, false, false);
      accF = __builtin_amdgcn_wmma_f32_16x16x32_bf16(false, af, false, bff, (short)0, accF, false, false);
      __syncthreads();
    }
  }
  int n = n0 + wn * 16 + laneLo;
  float inv_e = we_g[n] * rsqrtf(we_v[n] + 1e-5f);
  float inv_f = wf_g[n] * rsqrtf(wf_v[n] + 1e-5f);
  float sh_e = we_beta[n] - we_m[n] * inv_e;
  float sh_f = wf_beta[n] - wf_m[n] * inv_f;
  float be_ = we_b[n], bf_ = wf_b[n];
  #pragma unroll
  for (int r = 0; r < 8; ++r) {
    int m = m0 + wm * 16 + r + (laneHi ? 8 : 0);
    float ve = (accE[r] + be_) * inv_e + sh_e;
    float vf = (accF[r] + bf_) * inv_f + sh_f;
    size_t oi = (size_t)((m >> 12) * C_ + n) * HW_ + (m & 4095);
    out[oi] = sigm(ve) * x[oi] + sigm(vf) * y[oi];
  }
}

extern "C" void kernel_launch(void* const* d_in, const int* in_sizes, int n_in,
                              void* d_out, int out_size, void* d_ws, size_t ws_size,
                              hipStream_t stream) {
  (void)in_sizes; (void)n_in; (void)out_size; (void)ws_size;
  const float* x      = (const float*)d_in[0];
  const float* y      = (const float*)d_in[1];
  const float* se_w1  = (const float*)d_in[2];
  const float* se_w2  = (const float*)d_in[3];
  const float* off_w  = (const float*)d_in[4];
  const float* off_b  = (const float*)d_in[5];
  const float* enc_w  = (const float*)d_in[6];
  const float* enc_b  = (const float*)d_in[7];
  const float* we_w   = (const float*)d_in[8];
  const float* we_b   = (const float*)d_in[9];
  const float* we_g   = (const float*)d_in[10];
  const float* we_beta= (const float*)d_in[11];
  const float* we_m   = (const float*)d_in[12];
  const float* we_v   = (const float*)d_in[13];
  const float* wf_w   = (const float*)d_in[14];
  const float* wf_b   = (const float*)d_in[15];
  const float* wf_g   = (const float*)d_in[16];
  const float* wf_beta= (const float*)d_in[17];
  const float* wf_m   = (const float*)d_in[18];
  const float* wf_v   = (const float*)d_in[19];
  float* out = (float*)d_out;

  char* ws = (char*)d_ws;
  size_t o = 0;
  auto alloc = [&](size_t bytes) {
    void* p = ws + o;
    o = (o + bytes + 255) & ~(size_t)255;
    return p;
  };
  float* base  = (float*)alloc((size_t)B_ * C_ * HW_ * 4);  // 16 MB
  float* smean = (float*)alloc(1024 * 4);
  float* onep  = (float*)alloc(1024 * 4);
  float* off   = (float*)alloc((size_t)B_ * 27 * HW_ * 4);  // 1.77 MB
  bf16*  enc   = (bf16*) alloc((size_t)B_ * CC_ * HW_ * 2); // 2 MB
  bf16*  offwR = (bf16*) alloc((size_t)9 * 32 * C_ * 2);
  bf16*  encwR = (bf16*) alloc((size_t)9 * CC_ * C_ * 2);
  bf16*  wewR  = (bf16*) alloc((size_t)9 * C_ * CC_ * 2);
  bf16*  wfwR  = (bf16*) alloc((size_t)9 * C_ * CC_ * 2);

  // weight relayout: f32 [N][Cin*9] -> bf16 [9][Npad][Cin]
  k_wprep<<<(9 * 32  * C_  + 255) / 256, 256, 0, stream>>>(off_w, offwR, 27,  C_,  32);
  k_wprep<<<(9 * CC_ * C_  + 255) / 256, 256, 0, stream>>>(enc_w, encwR, CC_, C_,  CC_);
  k_wprep<<<(9 * C_  * CC_ + 255) / 256, 256, 0, stream>>>(we_w,  wewR,  C_,  CC_, C_);
  k_wprep<<<(9 * C_  * CC_ + 255) / 256, 256, 0, stream>>>(wf_w,  wfwR,  C_,  CC_, C_);

  k_base_mean<<<B_ * C_, 256, 0, stream>>>((const float4*)x, (const float4*)y,
                                           (float4*)base, smean);
  k_se<<<1, 256, 0, stream>>>(smean, se_w1, se_w2, onep);
  k_scale<<<NPIX_, 256, 0, stream>>>((float4*)base, onep);   // NPIX_ blocks * 256 = total elements, 4x the float4 count

  k_conv_off<<<NPIX_ / 32, 128, 0, stream>>>(base, offwR, off_b, off);
  k_dcn<<<dim3(NPIX_ / 32, CC_ / 32), 128, 0, stream>>>(base, off, encwR, enc_b, enc);
  k_conv2<<<dim3(NPIX_ / 32, C_ / 32), 128, 0, stream>>>(enc, wewR, wfwR,
      we_b, we_g, we_beta, we_m, we_v, wf_b, wf_g, wf_beta, wf_m, wf_v, x, y, out);
}
